// Multiheads_197568495820
// MI455X (gfx1250) — matrix operand order repaired and checked by reading, not compiled, not run on hardware
//
#include <hip/hip_runtime.h>
#include <hip/hip_bf16.h>

// ---------------------------------------------------------------------------
// Problem sizes (fixed by the reference)
// ---------------------------------------------------------------------------
constexpr int B_  = 4;
constexpr int L_  = 2048;
constexpr int D_  = 1024;
constexpr int H_  = 16;
constexpr int HS_ = 64;
constexpr int DATT_ = H_ * HS_;          // 1024

typedef __attribute__((ext_vector_type(16))) __bf16 v16bf;
typedef __attribute__((ext_vector_type(8)))  float  v8f;
typedef unsigned short ushort_t;
typedef unsigned int   uint_t;

// Fragment = 16 bf16 values per lane = 8 VGPRs = 2 x b128 loads.
union Frag {
    v16bf v;
    uint4 q[2];
};

union Pack8 {
    ushort_t u[8];
    uint4    q;
};

__device__ inline ushort_t f2bf(float f) {
    uint_t u = __float_as_uint(f);
    uint_t r = u + 0x7FFFu + ((u >> 16) & 1u);   // round-to-nearest-even
    return (ushort_t)(r >> 16);
}

__device__ inline v8f wmma_bf16(v16bf a, v16bf b, v8f c) {
    // (neg_a, A, neg_b, B, c_mod, C, reuse_a, reuse_b)
    return __builtin_amdgcn_wmma_f32_16x16x32_bf16(
        false, a, false, b, (short)0, c, false, false);
}

// ---------------------------------------------------------------------------
// Kernel 1: fp32 -> bf16 conversion (grid-stride)
// ---------------------------------------------------------------------------
__global__ void cvt_f32_bf16(const float* __restrict__ src,
                             ushort_t* __restrict__ dst, int n) {
    int i = blockIdx.x * blockDim.x + threadIdx.x;
    int stride = gridDim.x * blockDim.x;
    for (; i < n; i += stride) dst[i] = f2bf(src[i]);
}

// ---------------------------------------------------------------------------
// Kernel 2: per-head projection  Out[b,h,l,hs] = sum_d X[b,l,d] * W[h,hs,d]
// One wave computes a 16(l) x 64(hs) tile. K-loop over D in steps of 32.
// transposed==0: Out layout [B,H,L,HS]  (used for Q, K)
// transposed==1: Out layout [B,H,HS,L]  (used for V: ready as P*V B-operand)
// ---------------------------------------------------------------------------
__global__ void proj_kernel(const ushort_t* __restrict__ X,
                            const ushort_t* __restrict__ W,
                            ushort_t* __restrict__ Out,
                            int transposed) {
    const int wave = blockIdx.x * (blockDim.x >> 5) + (threadIdx.x >> 5);
    const int lane = threadIdx.x & 31;
    const int half = lane >> 4;       // 0: lanes 0-15, 1: lanes 16-31
    const int ln   = lane & 15;

    const int mt = wave % (L_ / 16);
    const int bh = wave / (L_ / 16);
    const int b  = bh / H_;
    const int h  = bh % H_;
    const int l0 = mt * 16;

    const ushort_t* arow = X + ((size_t)b * L_ + l0 + ln) * D_;   // A row (lane)
    const ushort_t* Wh   = W + (size_t)h * HS_ * D_;

    v8f acc[4] = {v8f{}, v8f{}, v8f{}, v8f{}};

    for (int kk = 0; kk < D_; kk += 32) {
        Frag a;
        a.q[0] = *(const uint4*)(arow + kk + half * 8);        // K = kk+{0..7 | 8..15}
        a.q[1] = *(const uint4*)(arow + kk + 16 + half * 8);   // K = kk+16+{...}
#pragma unroll
        for (int t = 0; t < 4; ++t) {
            const ushort_t* bcol = Wh + (size_t)(t * 16 + ln) * D_ + kk + half * 8;
            Frag bf;
            bf.q[0] = *(const uint4*)(bcol);
            bf.q[1] = *(const uint4*)(bcol + 16);
            acc[t] = wmma_bf16(a.v, bf.v, acc[t]);
        }
    }

    if (!transposed) {
        // Out[b][h][l][hs]; C-layout: VGPR r -> M = r + 8*half, lane -> N = ln
        const size_t base = ((size_t)b * H_ + h) * L_;
#pragma unroll
        for (int t = 0; t < 4; ++t) {
            const int hs = t * 16 + ln;
#pragma unroll
            for (int r = 0; r < 8; ++r) {
                const int l = l0 + r + 8 * half;
                Out[(base + l) * HS_ + hs] = f2bf(acc[t][r]);
            }
        }
    } else {
        // Out[b][h][hs][l]; M contiguous per lane -> one packed b128 store/tile
        const size_t base = ((size_t)b * H_ + h) * HS_;
#pragma unroll
        for (int t = 0; t < 4; ++t) {
            const int hs = t * 16 + ln;
            Pack8 p;
#pragma unroll
            for (int r = 0; r < 8; ++r) p.u[r] = f2bf(acc[t][r]);
            *(uint4*)(Out + (base + hs) * (size_t)L_ + l0 + 8 * half) = p.q;
        }
    }
}

// ---------------------------------------------------------------------------
// Kernel 3: flash attention per (b,h). One wave owns 16 queries, streams keys
// in chunks of 32. S stays in accumulators; P goes through 1KB LDS to flip
// C-layout -> A-layout for the P*V WMMAs. V is pre-transposed [B,H,HS,L].
// Output written directly in concat layout [B,L,DATT] (bf16).
// ---------------------------------------------------------------------------
__global__ void flash_attn(const ushort_t* __restrict__ Qp,
                           const ushort_t* __restrict__ Kp,
                           const ushort_t* __restrict__ Vt,
                           const int* __restrict__ msk,
                           ushort_t* __restrict__ AO) {
    __shared__ __align__(16) ushort_t plds[8][16 * 32];   // per-wave P tile (bf16)

    const int wslot = threadIdx.x >> 5;
    const int wave  = blockIdx.x * (blockDim.x >> 5) + wslot;
    const int lane  = threadIdx.x & 31;
    const int half  = lane >> 4;
    const int ln    = lane & 15;

    const int qt = wave % (L_ / 16);
    const int bh = wave / (L_ / 16);
    const int b  = bh / H_;
    const int h  = bh % H_;
    const int q0 = qt * 16;

    // Q A-fragments (held for the whole key loop): rows q0+ln, d = 0..63
    const ushort_t* qrow = Qp + (((size_t)b * H_ + h) * L_ + q0 + ln) * HS_;
    Frag qa0, qa1;
    qa0.q[0] = *(const uint4*)(qrow + half * 8);
    qa0.q[1] = *(const uint4*)(qrow + 16 + half * 8);
    qa1.q[0] = *(const uint4*)(qrow + 32 + half * 8);
    qa1.q[1] = *(const uint4*)(qrow + 48 + half * 8);

    const ushort_t* Kh = Kp + (((size_t)b * H_ + h) * L_) * HS_;
    const ushort_t* Vh = Vt + (((size_t)b * H_ + h) * HS_) * L_;
    const int*      mb = msk + (size_t)b * L_;

    float mrun[8], lrun[8];
#pragma unroll
    for (int r = 0; r < 8; ++r) { mrun[r] = -1e30f; lrun[r] = 0.0f; }
    v8f o[4] = {v8f{}, v8f{}, v8f{}, v8f{}};

    constexpr float scale = 0.125f;   // HS^-0.5 = 1/8

    for (int kb = 0; kb < L_; kb += 32) {
        // ---- S = Q * K^T for 2 key tiles of 16 ----
        v8f s[2];
#pragma unroll
        for (int t = 0; t < 2; ++t) {
            const ushort_t* krow = Kh + (size_t)(kb + t * 16 + ln) * HS_;
            Frag k0, k1;
            k0.q[0] = *(const uint4*)(krow + half * 8);
            k0.q[1] = *(const uint4*)(krow + 16 + half * 8);
            k1.q[0] = *(const uint4*)(krow + 32 + half * 8);
            k1.q[1] = *(const uint4*)(krow + 48 + half * 8);
            v8f c = v8f{};
            c = wmma_bf16(qa0.v, k0.v, c);
            c = wmma_bf16(qa1.v, k1.v, c);
            const int key = kb + t * 16 + ln;       // this lane's column
            const bool on = mb[key] != 0;
#pragma unroll
            for (int r = 0; r < 8; ++r)
                s[t][r] = on ? c[r] * scale : -1e9f;
        }

        // ---- online softmax stats (rows live across 16-lane halves) ----
        float csc[8];
#pragma unroll
        for (int r = 0; r < 8; ++r) {
            float v = fmaxf(s[0][r], s[1][r]);
            v = fmaxf(v, __shfl_xor(v, 1, 32));
            v = fmaxf(v, __shfl_xor(v, 2, 32));
            v = fmaxf(v, __shfl_xor(v, 4, 32));
            v = fmaxf(v, __shfl_xor(v, 8, 32));
            const float mnew = fmaxf(mrun[r], v);
            csc[r] = __expf(mrun[r] - mnew);
            const float p0 = __expf(s[0][r] - mnew);
            const float p1 = __expf(s[1][r] - mnew);
            s[0][r] = p0; s[1][r] = p1;
            float ps = p0 + p1;
            ps += __shfl_xor(ps, 1, 32);
            ps += __shfl_xor(ps, 2, 32);
            ps += __shfl_xor(ps, 4, 32);
            ps += __shfl_xor(ps, 8, 32);
            lrun[r] = lrun[r] * csc[r] + ps;
            mrun[r] = mnew;
        }
#pragma unroll
        for (int t2 = 0; t2 < 4; ++t2)
#pragma unroll
            for (int r = 0; r < 8; ++r) o[t2][r] *= csc[r];

        // ---- P: C-layout -> A-layout via per-wave LDS tile ----
        ushort_t* pl = plds[wslot];
#pragma unroll
        for (int r = 0; r < 8; ++r) {
            const int m = r + 8 * half;
            pl[m * 32 + ln]      = f2bf(s[0][r]);
            pl[m * 32 + 16 + ln] = f2bf(s[1][r]);
        }
        asm volatile("s_wait_dscnt 0" ::: "memory");   // LDS in-order per wave; belt&braces
        Frag pa;   // A-fragment: row = ln, K = keys 0..31 of this chunk
        pa.q[0] = *(const uint4*)(pl + ln * 32 + half * 8);
        pa.q[1] = *(const uint4*)(pl + ln * 32 + 16 + half * 8);

        // ---- O += P * V (V pre-transposed: contiguous keys per d-row) ----
#pragma unroll
        for (int t2 = 0; t2 < 4; ++t2) {
            const ushort_t* vrow = Vh + (size_t)(t2 * 16 + ln) * L_ + kb + half * 8;
            Frag vb;
            vb.q[0] = *(const uint4*)(vrow);
            vb.q[1] = *(const uint4*)(vrow + 16);
            o[t2] = wmma_bf16(pa.v, vb.v, o[t2]);
        }
    }

    // ---- normalize + store concat-layout bf16 output ----
    float inv[8];
#pragma unroll
    for (int r = 0; r < 8; ++r) inv[r] = 1.0f / lrun[r];
#pragma unroll
    for (int t2 = 0; t2 < 4; ++t2) {
        const int col = h * HS_ + t2 * 16 + ln;
#pragma unroll
        for (int r = 0; r < 8; ++r) {
            const int m = r + 8 * half;
            AO[((size_t)b * L_ + q0 + m) * DATT_ + col] = f2bf(o[t2][r] * inv[r]);
        }
    }
}

// ---------------------------------------------------------------------------
// Kernel 4: output projection  Y[m, o] = sum_c Xa[m, c] * Wp[o, c] + bp[o]
// M = B*L = 8192, N = K = 1024. One wave -> 16x64 tile. fp32 output.
// ---------------------------------------------------------------------------
__global__ void out_proj(const ushort_t* __restrict__ Xa,
                         const ushort_t* __restrict__ Wp,
                         const float* __restrict__ bp,
                         float* __restrict__ Y) {
    const int wave = blockIdx.x * (blockDim.x >> 5) + (threadIdx.x >> 5);
    const int lane = threadIdx.x & 31;
    const int half = lane >> 4;
    const int ln   = lane & 15;

    const int nt = wave % (DATT_ / 64);
    const int mt = wave / (DATT_ / 64);
    const int m0 = mt * 16;
    const int n0 = nt * 64;

    const ushort_t* arow = Xa + ((size_t)m0 + ln) * DATT_;
    v8f acc[4] = {v8f{}, v8f{}, v8f{}, v8f{}};

    for (int kk = 0; kk < DATT_; kk += 32) {
        Frag a;
        a.q[0] = *(const uint4*)(arow + kk + half * 8);
        a.q[1] = *(const uint4*)(arow + kk + 16 + half * 8);
#pragma unroll
        for (int t = 0; t < 4; ++t) {
            const ushort_t* bcol = Wp + (size_t)(n0 + t * 16 + ln) * DATT_ + kk + half * 8;
            Frag bf;
            bf.q[0] = *(const uint4*)(bcol);
            bf.q[1] = *(const uint4*)(bcol + 16);
            acc[t] = wmma_bf16(a.v, bf.v, acc[t]);
        }
    }

#pragma unroll
    for (int t = 0; t < 4; ++t) {
        const int col = n0 + t * 16 + ln;
        const float bias = bp[col];
#pragma unroll
        for (int r = 0; r < 8; ++r) {
            const int m = m0 + r + 8 * half;
            Y[(size_t)m * DATT_ + col] = acc[t][r] + bias;
        }
    }
}

// ---------------------------------------------------------------------------
// Host launcher
// ---------------------------------------------------------------------------
extern "C" void kernel_launch(void* const* d_in, const int* in_sizes, int n_in,
                              void* d_out, int out_size, void* d_ws, size_t ws_size,
                              hipStream_t stream) {
    (void)in_sizes; (void)n_in; (void)out_size; (void)ws_size;

    const float* idx_k = (const float*)d_in[0];
    const float* idx_q = (const float*)d_in[1];
    const float* idx_v = (const float*)d_in[2];
    const int*   msk   = (const int*)  d_in[3];
    const float* WK    = (const float*)d_in[4];
    const float* WQ    = (const float*)d_in[5];
    const float* WV    = (const float*)d_in[6];
    const float* Wp    = (const float*)d_in[7];
    const float* bp    = (const float*)d_in[8];
    float*       out   = (float*)d_out;

    constexpr size_t NX = (size_t)B_ * L_ * D_;      // 8,388,608
    constexpr size_t NW = (size_t)H_ * HS_ * D_;     // 1,048,576

    ushort_t* w   = (ushort_t*)d_ws;
    ushort_t* Xk  = w; w += NX;
    ushort_t* Xq  = w; w += NX;
    ushort_t* Xv  = w; w += NX;
    ushort_t* WKb = w; w += NW;
    ushort_t* WQb = w; w += NW;
    ushort_t* WVb = w; w += NW;
    ushort_t* Wpb = w; w += NW;
    ushort_t* Kp  = w; w += NX;   // [B,H,L,HS]
    ushort_t* Qp  = w; w += NX;   // [B,H,L,HS]
    ushort_t* Vtp = w; w += NX;   // [B,H,HS,L]  (transposed)
    ushort_t* AO  = w; w += NX;   // [B,L,DATT]

    // 1) fp32 -> bf16 conversions
    cvt_f32_bf16<<<2048, 256, 0, stream>>>(idx_k, Xk, (int)NX);
    cvt_f32_bf16<<<2048, 256, 0, stream>>>(idx_q, Xq, (int)NX);
    cvt_f32_bf16<<<2048, 256, 0, stream>>>(idx_v, Xv, (int)NX);
    cvt_f32_bf16<<<512,  256, 0, stream>>>(WK, WKb, (int)NW);
    cvt_f32_bf16<<<512,  256, 0, stream>>>(WQ, WQb, (int)NW);
    cvt_f32_bf16<<<512,  256, 0, stream>>>(WV, WVb, (int)NW);
    cvt_f32_bf16<<<512,  256, 0, stream>>>(Wp, Wpb, (int)NW);

    // 2) per-head projections: B*H*(L/16) = 8192 waves, 8 waves/block
    proj_kernel<<<1024, 256, 0, stream>>>(Xq, WQb, Qp,  0);
    proj_kernel<<<1024, 256, 0, stream>>>(Xk, WKb, Kp,  0);
    proj_kernel<<<1024, 256, 0, stream>>>(Xv, WVb, Vtp, 1);

    // 3) flash attention: 8192 waves
    flash_attn<<<1024, 256, 0, stream>>>(Qp, Kp, Vtp, msk, AO);

    // 4) output projection: (8192/16)*(1024/64) = 8192 waves
    out_proj<<<1024, 256, 0, stream>>>(AO, Wpb, bp, out);
}
